// ConstrainedRNN_2851858285110
// MI455X (gfx1250) — matrix operand order repaired and checked by reading, not compiled, not run on hardware
//
#include <hip/hip_runtime.h>
#include <hip/hip_bf16.h>

#define B_ 128
#define T_ 1024
#define I_ 256
#define H_ 512
#define C_ 32

typedef __attribute__((ext_vector_type(16))) __bf16        v16bf;
typedef __attribute__((ext_vector_type(8)))  float         v8f;
typedef __attribute__((ext_vector_type(4)))  unsigned int  u32x4;
typedef __attribute__((ext_vector_type(4)))  float         f32x4;

union Frag { v16bf v; u32x4 q[2]; };

static __device__ __forceinline__ v8f wmma_bf16(v16bf a, v16bf b, v8f c) {
    // emits v_wmma_f32_16x16x32_bf16 (args: neg_a, A, neg_b, B, c_mod, C, reuse_a, reuse_b)
    return __builtin_amdgcn_wmma_f32_16x16x32_bf16(false, a, false, b, (short)0, c, false, false);
}

// ---------------------------------------------------------------------------
// Kernel 1: convert W_hh and W_ih (f32 row-major) to bf16 in a WMMA-fragment-
// swizzled layout: whh_sw[((nt*16+kk)*32+lane)*16+e], so each lane's 16-element
// B-operand fragment is one contiguous 32B chunk (coalesced 1KB/wave loads).
// B k order used here (contiguous, not AMD's table order): lanes 0-15 hold K=0..15, lanes 16-31 hold
// K=16..31; element e -> K = kk*32 + 16*(lane>>4) + e; column = nt*16 + (lane&15).
// ---------------------------------------------------------------------------
__global__ void convert_weights(const float* __restrict__ Whh,
                                const float* __restrict__ Wih,
                                __bf16* __restrict__ whh_sw,
                                __bf16* __restrict__ wih_sw) {
    int idx = blockIdx.x * blockDim.x + threadIdx.x;
    const int NHH = H_ * H_;
    if (idx < NHH) {
        int e = idx & 15, lane = (idx >> 4) & 31, kk = (idx >> 9) & 15, nt = idx >> 13;
        int col = nt * 16 + (lane & 15);
        int k   = kk * 32 + 16 * (lane >> 4) + e;
        whh_sw[idx] = (__bf16)Whh[col * H_ + k];
    } else {
        int j = idx - NHH;
        if (j < H_ * I_) {
            int e = j & 15, lane = (j >> 4) & 31, kk = (j >> 9) & 7, nt = j >> 12;
            int col = nt * 16 + (lane & 15);
            int k   = kk * 32 + 16 * (lane >> 4) + e;
            wih_sw[j] = (__bf16)Wih[col * I_ + k];
        }
    }
}

// ---------------------------------------------------------------------------
// Kernel 2: cp[b][h] = constraints[b] . W_ch[h] + b_ch[h] + b_hh[h]  (tiny)
// ---------------------------------------------------------------------------
__global__ void compute_cp(const float* __restrict__ cons,
                           const float* __restrict__ Wch,
                           const float* __restrict__ bch,
                           const float* __restrict__ bhh,
                           float* __restrict__ cp) {
    int idx = blockIdx.x * blockDim.x + threadIdx.x;
    if (idx >= B_ * H_) return;
    int b = idx >> 9, h = idx & (H_ - 1);
    float s = 0.0f;
#pragma unroll
    for (int c = 0; c < C_; ++c) s += cons[b * C_ + c] * Wch[h * C_ + c];
    cp[idx] = s + bch[h] + bhh[h];
}

// ---------------------------------------------------------------------------
// Kernel 3: xp = x @ W_ih^T + b_ih, written straight into d_out (B,T,H layout:
// row m = b*T + t maps to out + m*H). One block per 16-row M tile; x tile is
// staged f32->bf16 into LDS once (K=I=256 fits entirely); 8 waves each own a
// 64-column slice (4 accumulators, A-fragment reuse x4).
// ---------------------------------------------------------------------------
__global__ __launch_bounds__(256) void xp_gemm(const float* __restrict__ x,
                                               const __bf16* __restrict__ wih_sw,
                                               const float* __restrict__ bih,
                                               float* __restrict__ out) {
    __shared__ __align__(16) __bf16 xs[16 * 264];   // 264 = 256 + 8 pad (bank-conflict-free)
    const int tid = threadIdx.x;
    const int M0  = blockIdx.x * 16;

    { // cooperative stage: 256 threads x 16 elements
        int row = tid >> 4;
        int col = (tid & 15) * 16;
        const f32x4* xp4 = reinterpret_cast<const f32x4*>(x + (M0 + row) * I_ + col);
        f32x4 f0 = xp4[0], f1 = xp4[1], f2 = xp4[2], f3 = xp4[3];
        float fv[16] = {f0.x,f0.y,f0.z,f0.w, f1.x,f1.y,f1.z,f1.w,
                        f2.x,f2.y,f2.z,f2.w, f3.x,f3.y,f3.z,f3.w};
        Frag fr;
#pragma unroll
        for (int i = 0; i < 16; ++i) fr.v[i] = (__bf16)fv[i];
        u32x4* dst = reinterpret_cast<u32x4*>(xs + row * 264 + col);
        dst[0] = fr.q[0]; dst[1] = fr.q[1];
    }
    __syncthreads();

    const int wave = tid >> 5, lane = tid & 31;
    const int nlo = lane & 15, hh = lane >> 4;

    v8f acc[4];
    const v8f zero = {0.f,0.f,0.f,0.f,0.f,0.f,0.f,0.f};
#pragma unroll
    for (int j = 0; j < 4; ++j) acc[j] = zero;

    const u32x4* xs4 = reinterpret_cast<const u32x4*>(xs);
    const u32x4* w4  = reinterpret_cast<const u32x4*>(wih_sw);

#pragma unroll
    for (int kk = 0; kk < 8; ++kk) {
        // A fragment (this code's contiguous k order, same as B: lane half selects K 16h..16h+15)
        int aoff = nlo * 264 + kk * 32 + 16 * hh;          // elements, /8 -> u32x4 index
        Frag fa; fa.q[0] = xs4[aoff >> 3]; fa.q[1] = xs4[(aoff >> 3) + 1];
#pragma unroll
        for (int j = 0; j < 4; ++j) {
            int nt = wave * 4 + j;
            const u32x4* wp = w4 + ((((nt * 8 + kk) * 32) + lane) << 1);
            Frag fb; fb.q[0] = wp[0]; fb.q[1] = wp[1];
            acc[j] = wmma_bf16(fa.v, fb.v, acc[j]);
        }
    }

    const int mg = hh * 8;
#pragma unroll
    for (int j = 0; j < 4; ++j) {
        int col = wave * 64 + j * 16 + nlo;
        float bv = bih[col];
#pragma unroll
        for (int r = 0; r < 8; ++r) {
            int row = M0 + mg + r;
            out[row * H_ + col] = acc[j][r] + bv;
        }
    }
}

// ---------------------------------------------------------------------------
// Kernel 4: sequential scan. 8 blocks (16 batch rows each), 32 waves; h tile
// (16x512 bf16) lives in LDS (padded stride 520 -> conflict-free ds_load_b128).
// Per step, wave w computes columns [16w,16w+16) via 16 chained bf16 WMMAs
// (K=512), then tanh + masking, in-place overwrite of xp in d_out, writes
// `last` at t == len-1. No cross-workgroup sync needed (batch-parallel).
// ---------------------------------------------------------------------------
__global__ __launch_bounds__(1024) void rnn_scan(const float* __restrict__ cp,
                                                 const __bf16* __restrict__ whh_sw,
                                                 const int* __restrict__ lengths,
                                                 float* __restrict__ out) {
    __shared__ __align__(16) __bf16 hlds[16 * 520];
    const int tid = threadIdx.x;

    { // h0 = 0
        unsigned int* z = reinterpret_cast<unsigned int*>(hlds);
        for (int i = tid; i < (16 * 520 * 2) / 4; i += 1024) z[i] = 0u;
    }
    __syncthreads();

    const int wave = tid >> 5, lane = tid & 31;
    const int nlo = lane & 15, hh = lane >> 4, mg = hh * 8;
    const int B0 = blockIdx.x * 16;
    const int N0 = wave * 16;

    int   lenr[8];
    float cpr[8];
    int   obase[8];
    int   lbase[8];
#pragma unroll
    for (int r = 0; r < 8; ++r) {
        int b    = B0 + mg + r;
        lenr[r]  = lengths[b];
        cpr[r]   = cp[b * H_ + N0 + nlo];
        obase[r] = b * T_ * H_ + N0 + nlo;
        lbase[r] = B_ * T_ * H_ + b * H_ + N0 + nlo;
    }

    const u32x4* hs4 = reinterpret_cast<const u32x4*>(hlds);
    const u32x4* w4  = reinterpret_cast<const u32x4*>(whh_sw);
    const v8f zero = {0.f,0.f,0.f,0.f,0.f,0.f,0.f,0.f};

#pragma unroll 1
    for (int t = 0; t < T_; ++t) {
        v8f acc = zero;
#pragma unroll
        for (int kk = 0; kk < 16; ++kk) {
            int aoff = nlo * 520 + kk * 32 + 16 * hh;      // elements
            Frag fa; fa.q[0] = hs4[aoff >> 3]; fa.q[1] = hs4[(aoff >> 3) + 1];
            const u32x4* wp = w4 + ((((wave * 16 + kk) * 32) + lane) << 1);
            Frag fb; fb.q[0] = wp[0]; fb.q[1] = wp[1];
            acc = wmma_bf16(fa.v, fb.v, acc);
        }
        __syncthreads();  // all reads of h(t-1) done before anyone writes h(t)

#pragma unroll
        for (int r = 0; r < 8; ++r) {
            int   oidx = obase[r] + t * H_;
            float xpv  = out[oidx];
            float hn   = tanhf(acc[r] + cpr[r] + xpv);
            bool  act  = t < lenr[r];
            out[oidx]  = act ? hn : 0.0f;
            if (act) hlds[(mg + r) * 520 + N0 + nlo] = (__bf16)hn;
            if (t == lenr[r] - 1) out[lbase[r]] = hn;
        }
        __syncthreads();  // h(t) writes visible before next step's reads
    }
}

// ---------------------------------------------------------------------------
extern "C" void kernel_launch(void* const* d_in, const int* in_sizes, int n_in,
                              void* d_out, int out_size, void* d_ws, size_t ws_size,
                              hipStream_t stream) {
    const float* x       = (const float*)d_in[0];
    const float* cons    = (const float*)d_in[1];
    const int*   lengths = (const int*)d_in[2];
    const float* W_ih    = (const float*)d_in[3];
    const float* b_ih    = (const float*)d_in[4];
    const float* W_hh    = (const float*)d_in[5];
    const float* b_hh    = (const float*)d_in[6];
    const float* W_ch    = (const float*)d_in[7];
    const float* b_ch    = (const float*)d_in[8];
    float* out = (float*)d_out;

    // workspace layout (1 MB total)
    __bf16* whh_sw = (__bf16*)d_ws;                               // 512 KB
    __bf16* wih_sw = (__bf16*)((char*)d_ws + 524288);             // 256 KB
    float*  cp     = (float*) ((char*)d_ws + 786432);             // 256 KB

    convert_weights<<<1536, 256, 0, stream>>>(W_hh, W_ih, whh_sw, wih_sw);
    compute_cp<<<256, 256, 0, stream>>>(cons, W_ch, b_ch, b_hh, cp);
    xp_gemm<<<(B_ * T_) / 16, 256, 0, stream>>>(x, wih_sw, b_ih, out);
    rnn_scan<<<B_ / 16, 1024, 0, stream>>>(cp, whh_sw, lengths, out);
}
